// _MyAttentionBlock_38019050504199
// MI455X (gfx1250) — matrix operand order repaired and checked by reading, not compiled, not run on hardware
//
#include <hip/hip_runtime.h>
#include <math.h>

#define EMBD  1024
#define BATCH 8
#define SEQ   2048
#define ROWS  (BATCH * SEQ)

#define MT   128   // block tile M
#define NT   128   // block tile N
#define KTS  32    // K tile = one WMMA k-step
#define LDSK 40    // padded LDS row stride (halfwords) -> conflict-free b128 reads

typedef unsigned short u16;
typedef __attribute__((ext_vector_type(16))) __bf16 v16bf;
typedef __attribute__((ext_vector_type(8)))  float  v8f;
typedef int v4i_ __attribute__((__vector_size__(4 * sizeof(int))));

#if defined(__AMDGCN__) && __has_builtin(__builtin_amdgcn_global_load_async_to_lds_b128)
#define HAVE_ASYNC_LDS 1
#else
#define HAVE_ASYNC_LDS 0
#endif

#define GPTR(p) ((__attribute__((address_space(1))) v4i_*)(p))
#define LPTR(p) ((__attribute__((address_space(3))) v4i_*)(p))

__device__ __forceinline__ void wait_async0() {
#if defined(__AMDGCN__)
#if __has_builtin(__builtin_amdgcn_s_wait_asynccnt)
  __builtin_amdgcn_s_wait_asynccnt(0);
#else
  asm volatile("s_wait_asynccnt 0x0" ::: "memory");
#endif
#endif
}

__device__ __forceinline__ u16 f2bf(float f) {
  unsigned int u = __float_as_uint(f);
  u += 0x7FFFu + ((u >> 16) & 1u);   // round-to-nearest-even
  return (u16)(u >> 16);
}

struct FragBF { union { uint4 q[2]; v16bf v; }; };

// ------------------------------------------------ fused transpose + cast f32->bf16
// src: [Kd x Nd] f32 row-major   dst: [Nd x Kd] bf16 row-major
__global__ __launch_bounds__(256) void transcast_kernel(
    const float* __restrict__ src, u16* __restrict__ dst, int Kd, int Nd) {
  __shared__ u16 t[32][33];          // t[n_off][k_off]
  const int k0 = blockIdx.x * 32;
  const int n0 = blockIdx.y * 32;
  const int lx = threadIdx.x & 31;
  const int ly = threadIdx.x >> 5;   // 0..7
#pragma unroll
  for (int i = 0; i < 4; ++i) {
    int kk = ly + i * 8;
    t[lx][kk] = f2bf(src[(size_t)(k0 + kk) * Nd + n0 + lx]);
  }
  __syncthreads();
#pragma unroll
  for (int i = 0; i < 4; ++i) {
    int nn = ly + i * 8;
    dst[(size_t)(n0 + nn) * Kd + k0 + lx] = t[nn][lx];
  }
}

// ------------------------------------------------ layernorm -> bf16
__global__ __launch_bounds__(256) void layernorm_bf16_kernel(
    const float* __restrict__ x, const float* __restrict__ w,
    const float* __restrict__ b, u16* __restrict__ y) {
  const int row = blockIdx.x;
  const int tid = threadIdx.x;
  const float* xr = x + (size_t)row * EMBD;
  float v0 = xr[tid], v1 = xr[tid + 256], v2 = xr[tid + 512], v3 = xr[tid + 768];
  float s  = v0 + v1 + v2 + v3;
  float ss = v0 * v0 + v1 * v1 + v2 * v2 + v3 * v3;
  __shared__ float rs[256], rq[256];
  rs[tid] = s; rq[tid] = ss;
  __syncthreads();
  for (int o = 128; o > 0; o >>= 1) {
    if (tid < o) { rs[tid] += rs[tid + o]; rq[tid] += rq[tid + o]; }
    __syncthreads();
  }
  float mu  = rs[0] * (1.0f / EMBD);
  float var = rq[0] * (1.0f / EMBD) - mu * mu;
  float inv = rsqrtf(var + 1e-5f);
  u16* yr = y + (size_t)row * EMBD;
  yr[tid +   0] = f2bf((v0 - mu) * inv * w[tid +   0] + b[tid +   0]);
  yr[tid + 256] = f2bf((v1 - mu) * inv * w[tid + 256] + b[tid + 256]);
  yr[tid + 512] = f2bf((v2 - mu) * inv * w[tid + 512] + b[tid + 512]);
  yr[tid + 768] = f2bf((v3 - mu) * inv * w[tid + 768] + b[tid + 768]);
}

// ------------------------------------------------ row softmax (fp32 in, bf16 out)
__global__ __launch_bounds__(256) void softmax_kernel(
    const float* __restrict__ S, u16* __restrict__ P) {
  const int row = blockIdx.x;
  const int tid = threadIdx.x;
  const float* sr = S + (size_t)row * SEQ;
  float v[8];
  float m = -3.0e38f;
#pragma unroll
  for (int i = 0; i < 8; ++i) { v[i] = sr[tid + i * 256]; m = fmaxf(m, v[i]); }
  __shared__ float red[256];
  red[tid] = m;
  __syncthreads();
  for (int o = 128; o > 0; o >>= 1) {
    if (tid < o) red[tid] = fmaxf(red[tid], red[tid + o]);
    __syncthreads();
  }
  m = red[0];
  __syncthreads();
  float s = 0.f;
#pragma unroll
  for (int i = 0; i < 8; ++i) { v[i] = expf(v[i] - m); s += v[i]; }
  red[tid] = s;
  __syncthreads();
  for (int o = 128; o > 0; o >>= 1) {
    if (tid < o) red[tid] += red[tid + o];
    __syncthreads();
  }
  float inv = 1.0f / red[0];
  u16* pr = P + (size_t)row * SEQ;
#pragma unroll
  for (int i = 0; i < 8; ++i) pr[tid + i * 256] = f2bf(v[i] * inv);
}

// ------------------------------------------------ WMMA GEMM
// C[M x Nc] = A[M x K](bf16, row-major, ld=K) * B^T  with B stored [Nc x K](bf16, row stride=ldb)
// EPI: 0 = bf16 out, +bias
//      1 = f32 out, *scale
//      2 = f32 out, +resid
//      3 = bf16 out, +bias, exact-erf GELU
//      4 = f32 out, +bias, +resid
//      5 = bf16 out, +bias, TRANSPOSED store: C[col*M + row]
template <int EPI>
__global__ __launch_bounds__(256) void gemm_wmma_kernel(
    const u16* __restrict__ A, const u16* __restrict__ B,
    const float* __restrict__ bias, const float* __restrict__ resid,
    void* __restrict__ Cout, float scale,
    int M, int Nc, int K, int ldb, long sA, long sB, long sC) {
  __shared__ __align__(16) u16 As[2][MT][LDSK];
  __shared__ __align__(16) u16 Bs[2][NT][LDSK];

  const int tid  = threadIdx.x;
  const int lane = tid & 31;
  const int wid  = tid >> 5;
  const int wm   = wid & 1;    // 2 waves along M (64 rows each)
  const int wn   = wid >> 1;   // 4 waves along N (32 cols each)
  const int hf   = lane >> 4;  // lane half (WMMA layout)
  const int l15  = lane & 15;

  const int brow = blockIdx.y * MT;
  const int bcol = blockIdx.x * NT;
  const int z    = blockIdx.z;

  A += (size_t)z * sA;
  B += (size_t)z * sB;
  const size_t coff = (size_t)z * sC;

  const int nkt = K / KTS;

  const int arow = tid >> 1;          // 0..127
  const int acol = (tid & 1) * 16;    // 0 or 16

  const u16* agp = A + (size_t)(brow + arow) * K   + acol;
  const u16* bgp = B + (size_t)(bcol + arow) * ldb + acol;

#if HAVE_ASYNC_LDS
  // CDNA5 async DMA global->LDS (ASYNCcnt); INST_OFFSET applies to both addresses
  auto fill = [&](int kt, int buf) {
    const u16* ag = agp + kt * KTS;
    const u16* bg = bgp + kt * KTS;
    u16* la = &As[buf][arow][acol];
    u16* lb = &Bs[buf][arow][acol];
    __builtin_amdgcn_global_load_async_to_lds_b128(GPTR(ag), LPTR(la), 0, 0);
    __builtin_amdgcn_global_load_async_to_lds_b128(GPTR(ag), LPTR(la), 16, 0);
    __builtin_amdgcn_global_load_async_to_lds_b128(GPTR(bg), LPTR(lb), 0, 0);
    __builtin_amdgcn_global_load_async_to_lds_b128(GPTR(bg), LPTR(lb), 16, 0);
  };
#else
  uint4 ra0, ra1, rb0, rb1;
  auto gload = [&](int kt) {
    const u16* ag = agp + kt * KTS;
    const u16* bg = bgp + kt * KTS;
    ra0 = *(const uint4*)ag;
    ra1 = *(const uint4*)(ag + 8);
    rb0 = *(const uint4*)bg;
    rb1 = *(const uint4*)(bg + 8);
  };
  auto sstore = [&](int buf) {
    u16* da = &As[buf][arow][acol];
    u16* db = &Bs[buf][arow][acol];
    *(uint4*)da = ra0; *(uint4*)(da + 8) = ra1;
    *(uint4*)db = rb0; *(uint4*)(db + 8) = rb1;
  };
#endif

  v8f acc[4][2];
  const v8f vz = {0.f, 0.f, 0.f, 0.f, 0.f, 0.f, 0.f, 0.f};
#pragma unroll
  for (int mi = 0; mi < 4; ++mi)
#pragma unroll
    for (int ni = 0; ni < 2; ++ni) acc[mi][ni] = vz;

#if HAVE_ASYNC_LDS
  fill(0, 0);
  wait_async0();
  __syncthreads();
#else
  gload(0);
  sstore(0);
  __syncthreads();
#endif

  for (int kt = 0; kt < nkt; ++kt) {
    const int cur = kt & 1;
#if HAVE_ASYNC_LDS
    if (kt + 1 < nkt) fill(kt + 1, cur ^ 1);   // DMA overlaps the WMMAs below
#else
    if (kt + 1 < nkt) gload(kt + 1);
#endif

    // fragment loads: A 16x32 bf16 layout (ISA 7.12.2), B 32x16 from [n][k] LDS tile
    FragBF fa[4], fb[2];
#pragma unroll
    for (int mi = 0; mi < 4; ++mi) {
      const u16* p = &As[cur][wm * 64 + mi * 16 + l15][hf * 8];
      fa[mi].q[0] = *(const uint4*)p;        // K = hf*8 .. hf*8+7
      fa[mi].q[1] = *(const uint4*)(p + 16); // K = 16+hf*8 .. 16+hf*8+7
    }
#pragma unroll
    for (int ni = 0; ni < 2; ++ni) {
      const u16* p = &Bs[cur][wn * 32 + ni * 16 + l15][hf * 8];
      fb[ni].q[0] = *(const uint4*)p;        // K = hf*8 .. hf*8+7
      fb[ni].q[1] = *(const uint4*)(p + 16); // K = 16+hf*8 .. 16+hf*8+7
    }
#pragma unroll
    for (int mi = 0; mi < 4; ++mi)
#pragma unroll
      for (int ni = 0; ni < 2; ++ni)
        acc[mi][ni] = __builtin_amdgcn_wmma_f32_16x16x32_bf16(
            false, fa[mi].v, false, fb[ni].v, (short)0, acc[mi][ni], false, false);

#if HAVE_ASYNC_LDS
    wait_async0();
#else
    if (kt + 1 < nkt) sstore(cur ^ 1);
#endif
    __syncthreads();
  }

  // epilogue; C frag: lane(hf,n): vgpr r -> row = hf*8 + r, col = n
  const int colB = bcol + wn * 32 + l15;
  const int rowB = brow + wm * 64 + hf * 8;
#pragma unroll
  for (int mi = 0; mi < 4; ++mi) {
#pragma unroll
    for (int ni = 0; ni < 2; ++ni) {
      const int col = colB + ni * 16;
      float bv = 0.f;
      if (EPI == 0 || EPI == 3 || EPI == 4 || EPI == 5) bv = bias[col];
#pragma unroll
      for (int r = 0; r < 8; ++r) {
        const int row = rowB + mi * 16 + r;
        float v = acc[mi][ni][r];
        if (EPI == 1) v *= scale;
        v += bv;
        if (EPI == 3) v = 0.5f * v * (1.0f + erff(v * 0.70710678118654752f));
        if (EPI == 2 || EPI == 4) v += resid[coff + (size_t)row * Nc + col];
        if (EPI == 0 || EPI == 3)
          ((u16*)Cout)[coff + (size_t)row * Nc + col] = f2bf(v);
        else if (EPI == 5)
          ((u16*)Cout)[coff + (size_t)col * M + row] = f2bf(v);
        else
          ((float*)Cout)[coff + (size_t)row * Nc + col] = v;
      }
    }
  }
}

// ------------------------------------------------ launch
extern "C" void kernel_launch(void* const* d_in, const int* in_sizes, int n_in,
                              void* d_out, int out_size, void* d_ws, size_t ws_size,
                              hipStream_t stream) {
  (void)in_sizes; (void)n_in; (void)out_size; (void)ws_size;
  const float* x    = (const float*)d_in[0];
  const float* ln1w = (const float*)d_in[1];
  const float* ln1b = (const float*)d_in[2];
  const float* wq   = (const float*)d_in[3];
  const float* bq   = (const float*)d_in[4];
  const float* wk   = (const float*)d_in[5];
  const float* bk   = (const float*)d_in[6];
  const float* wv   = (const float*)d_in[7];
  const float* bv   = (const float*)d_in[8];
  const float* ln2w = (const float*)d_in[9];
  const float* ln2b = (const float*)d_in[10];
  const float* w1   = (const float*)d_in[11];
  const float* b1   = (const float*)d_in[12];
  const float* w2   = (const float*)d_in[13];
  const float* b2   = (const float*)d_in[14];
  float* out = (float*)d_out;

  char* wsp = (char*)d_ws;
  size_t off = 0;
  auto carve = [&](size_t bytes) -> void* {
    void* p = wsp + off;
    off += (bytes + 255) & ~(size_t)255;
    return p;
  };

  u16*   yb  = (u16*)carve((size_t)ROWS * EMBD * 2);
  u16*   qb  = (u16*)carve((size_t)ROWS * EMBD * 2);
  u16*   kb  = (u16*)carve((size_t)ROWS * EMBD * 2);
  u16*   vT  = (u16*)carve((size_t)ROWS * EMBD * 2);   // stored [EMBD x ROWS]
  float* S   = (float*)carve((size_t)BATCH * SEQ * SEQ * 4);
  u16*   Pb  = (u16*)carve((size_t)BATCH * SEQ * SEQ * 2);
  u16*   wqT = (u16*)carve((size_t)EMBD * EMBD * 2);
  u16*   wkT = (u16*)carve((size_t)EMBD * EMBD * 2);
  u16*   wvT = (u16*)carve((size_t)EMBD * EMBD * 2);
  u16*   w1T = (u16*)carve((size_t)EMBD * 2 * EMBD * 2);
  u16*   w2T = (u16*)carve((size_t)EMBD * 2 * EMBD * 2);
  u16*   hb  = yb;        // reuse: y dead after q/k/v
  u16*   h1b = (u16*)S;   // reuse: S dead after softmax

  dim3 blk(256);

  // transpose-cast weights to bf16 [out x in] so every GEMM B-tile fill is a b128 copy
  transcast_kernel<<<dim3(32, 32), blk, 0, stream>>>(wq, wqT, EMBD, EMBD);
  transcast_kernel<<<dim3(32, 32), blk, 0, stream>>>(wk, wkT, EMBD, EMBD);
  transcast_kernel<<<dim3(32, 32), blk, 0, stream>>>(wv, wvT, EMBD, EMBD);
  transcast_kernel<<<dim3(32, 64), blk, 0, stream>>>(w1, w1T, EMBD, 2 * EMBD);
  transcast_kernel<<<dim3(64, 32), blk, 0, stream>>>(w2, w2T, 2 * EMBD, EMBD);

  // y = LN1(x)
  layernorm_bf16_kernel<<<ROWS, blk, 0, stream>>>(x, ln1w, ln1b, yb);

  // q,k = y @ W + b ; v stored transposed [EMBD x ROWS]
  {
    dim3 g(EMBD / NT, ROWS / MT, 1);
    gemm_wmma_kernel<0><<<g, blk, 0, stream>>>(yb, wqT, bq, nullptr, qb, 1.f,
                                               ROWS, EMBD, EMBD, EMBD, 0, 0, 0);
    gemm_wmma_kernel<0><<<g, blk, 0, stream>>>(yb, wkT, bk, nullptr, kb, 1.f,
                                               ROWS, EMBD, EMBD, EMBD, 0, 0, 0);
    gemm_wmma_kernel<5><<<g, blk, 0, stream>>>(yb, wvT, bv, nullptr, vT, 1.f,
                                               ROWS, EMBD, EMBD, EMBD, 0, 0, 0);
  }
  // S = (q @ k^T) / sqrt(d)   (batched; k rows are already the [Nc x K] layout)
  {
    dim3 g(SEQ / NT, SEQ / MT, BATCH);
    gemm_wmma_kernel<1><<<g, blk, 0, stream>>>(qb, kb, nullptr, nullptr, S, 0.03125f,
                                               SEQ, SEQ, EMBD, EMBD,
                                               (long)SEQ * EMBD, (long)SEQ * EMBD, (long)SEQ * SEQ);
  }
  // P = softmax(S)
  softmax_kernel<<<ROWS, blk, 0, stream>>>(S, Pb);
  // out = x + P @ v   (batched; B = vT with row stride ROWS, batch offset SEQ)
  {
    dim3 g(EMBD / NT, SEQ / MT, BATCH);
    gemm_wmma_kernel<2><<<g, blk, 0, stream>>>(Pb, vT, nullptr, x, out, 1.f,
                                               SEQ, EMBD, SEQ, ROWS,
                                               (long)SEQ * SEQ, (long)SEQ, (long)SEQ * EMBD);
  }
  // h = LN2(out)
  layernorm_bf16_kernel<<<ROWS, blk, 0, stream>>>(out, ln2w, ln2b, hb);
  // h1 = gelu(h @ w1 + b1)
  {
    dim3 g(2 * EMBD / NT, ROWS / MT, 1);
    gemm_wmma_kernel<3><<<g, blk, 0, stream>>>(hb, w1T, b1, nullptr, h1b, 1.f,
                                               ROWS, 2 * EMBD, EMBD, EMBD, 0, 0, 0);
  }
  // out = out + h1 @ w2 + b2
  {
    dim3 g(EMBD / NT, ROWS / MT, 1);
    gemm_wmma_kernel<4><<<g, blk, 0, stream>>>(h1b, w2T, b2, out, out, 1.f,
                                               ROWS, EMBD, 2 * EMBD, 2 * EMBD, 0, 0, 0);
  }
}
